// Physics_Attention_16836271800865
// MI455X (gfx1250) — matrix operand order repaired and checked by reading, not compiled, not run on hardware
//
#include <hip/hip_runtime.h>

// ---------------------------------------------------------------------------
// Physics attention (slice tokens) for MI455X / gfx1250, bf16 WMMA pipeline.
// Layouts chosen so every WMMA fragment load is pair-contiguous (fuses to
// global_load_b128) and no kernel needs an LDS transpose.
// ---------------------------------------------------------------------------

#define BATCH   4
#define NTOK    32768
#define CDIM    256
#define HEADS   8
#define DHEAD   64
#define GSLICE  64
#define INNER   512          // HEADS * DHEAD
#define ATTN_SCALE 0.125f    // DHEAD^-0.5

#define POOL_CHUNKS 32
#define POOL_SPAN   (NTOK / POOL_CHUNKS)   // 1024 n-rows per pooling block

typedef __attribute__((ext_vector_type(16))) __bf16 v16bf;
typedef __attribute__((ext_vector_type(8)))  float  v8f;
typedef __attribute__((ext_vector_type(8)))  __bf16 v8bf;
typedef __attribute__((ext_vector_type(2)))  __bf16 bf16x2;

__device__ __forceinline__ v8f v8f_zero() {
  v8f z = {0.f, 0.f, 0.f, 0.f, 0.f, 0.f, 0.f, 0.f};
  return z;
}

// ---- WMMA fragment loaders (CDNA5 wave32 16-bit layouts, CDNA5 ISA) --
// A (16x32, MxK) from row-major [M][K] source: lane 0-15 -> M=lane, K pairs
__device__ __forceinline__ v16bf frag_a_rowmajor(const __bf16* base, int ld, int lane) {
  const int half = (lane >> 4) & 1;
  const int m    = lane & 15;
  v16bf a;
#pragma unroll
  for (int i = 0; i < 8; ++i) {
    const int k = (i < 4 ? 2 * i : 16 + 2 * (i - 4)) + half * 8;
    bf16x2 p = *(const bf16x2*)(base + (size_t)m * ld + k);
    a[2 * i]     = p[0];
    a[2 * i + 1] = p[1];
  }
  return a;
}

// B (32x16, KxN) from transposed [N][K] source:
//   VGPR v<4, lanes 0-15: K = {2v,2v+1}, v>=4: K = 16+{2(v-4),+1}, N=lane ; lanes 16-31: K += 8
__device__ __forceinline__ v16bf frag_b_colmajor(const __bf16* baseT, int ld, int lane) {
  const int half = (lane >> 4) & 1;
  const int n    = lane & 15;
  v16bf b;
#pragma unroll
  for (int i = 0; i < 8; ++i) {
    const int k = (i < 4 ? 2 * i : 16 + 2 * (i - 4)) + half * 8;
    bf16x2 p = *(const bf16x2*)(baseT + (size_t)n * ld + k);
    b[2 * i]     = p[0];
    b[2 * i + 1] = p[1];
  }
  return b;
}

__device__ __forceinline__ v8f wmma_bf16(v16bf a, v16bf b, v8f c) {
  return __builtin_amdgcn_wmma_f32_16x16x32_bf16(false, a, false, b,
                                                 (short)0, c, false, false);
}

// ---------------------------------------------------------------------------
// prep: f32 [K][N] -> bf16 transposed [N][K]
// ---------------------------------------------------------------------------
__global__ void convT_kernel(const float* __restrict__ src,
                             __bf16* __restrict__ dst, int K, int N) {
  const int idx = blockIdx.x * 256 + threadIdx.x;
  if (idx < K * N) {
    const int k = idx / N;
    const int n = idx - k * N;
    dst[(size_t)n * K + k] = (__bf16)src[idx];
  }
}

__global__ void zero_kernel(float* __restrict__ p, int count) {
  const int idx = blockIdx.x * 256 + threadIdx.x;
  if (idx < count) p[idx] = 0.f;
}

// ---------------------------------------------------------------------------
// K1: projections + per-head slice softmax.
//   one workgroup = 16 rows of n for one batch. 8 waves, 256 threads.
//   writes: fxT [B*H][D][N]   (packed b128 stores along n)
//           swN [B*H][N][G]   (row-contiguous, for K4 A-frags)
//           swT [B*H][G][N]   (staged via LDS, for K2 A-frags + norm)
// ---------------------------------------------------------------------------
__global__ __launch_bounds__(256)
void proj_softmax_kernel(const float* __restrict__ x,
                         const __bf16* __restrict__ wfxT,   // [INNER][CDIM]
                         const __bf16* __restrict__ wxT,    // [INNER][CDIM]
                         const __bf16* __restrict__ wslT,   // [GSLICE][DHEAD]
                         const float* __restrict__ bfx,
                         const float* __restrict__ bx,
                         const float* __restrict__ bsl,
                         const float* __restrict__ temperature,
                         __bf16* __restrict__ fxT,
                         __bf16* __restrict__ swN,
                         __bf16* __restrict__ swT)
{
  __shared__ union {
    struct {
      __bf16 xa[16][CDIM];     //  8 KB   (stage 1 input)
      __bf16 xmid[16][INNER];  // 16 KB   (stage 1 -> stage 2)
    } a;
    __bf16 sstage[HEADS * GSLICE][16];  // 16 KB (softmax -> swT store)
  } sm;
  __shared__ float logit[16][INNER];    // 32 KB

  const int b    = blockIdx.y;
  const int n0   = blockIdx.x * 16;
  const int tid  = threadIdx.x;
  const int wave = tid >> 5;
  const int lane = tid & 31;
  const int half = lane >> 4;
  const int ni   = lane & 15;

  // stage x tile as bf16
  for (int i = tid; i < 16 * CDIM; i += 256) {
    const int r = i >> 8;
    const int c = i & (CDIM - 1);
    sm.a.xa[r][c] = (__bf16)x[((size_t)b * NTOK + n0 + r) * CDIM + c];
  }
  __syncthreads();

  // ---- stage 1: [fx | x_mid] = x @ [W_fx | W_x], K = 256 ----
  // Hoist the 8 A-fragments (whole 16x256 x-tile) into registers once;
  // they are reused by all 8 output tiles this wave owns.
  v16bf afr[8];
#pragma unroll
  for (int s = 0; s < 8; ++s)
    afr[s] = frag_a_rowmajor(&sm.a.xa[0][0] + 32 * s, CDIM, lane);

  for (int t = 0; t < 8; ++t) {
    const int tile = wave + 8 * t;           // 0..63 (uniform per wave)
    const int col0 = (tile & 31) * 16;
    const bool is_x = tile >= 32;
    const __bf16* wT = is_x ? wxT : wfxT;
    v8f acc = v8f_zero();
#pragma unroll
    for (int s = 0; s < 8; ++s) {
      v16bf bf = frag_b_colmajor(wT + (size_t)col0 * CDIM + 32 * s, CDIM, lane);
      acc = wmma_bf16(afr[s], bf, acc);
    }
    const int col = col0 + ni;
    if (is_x) {
      const float bias = bx[col];
#pragma unroll
      for (int r = 0; r < 8; ++r)
        sm.a.xmid[r + 8 * half][col] = (__bf16)(acc[r] + bias);
    } else {
      // fxT[bh][d][n]: this lane owns fixed d, 8 consecutive n -> one b128 store
      const int h = col >> 6, d = col & 63;
      const float bias = bfx[col];
      v8bf pk;
#pragma unroll
      for (int r = 0; r < 8; ++r) pk[r] = (__bf16)(acc[r] + bias);
      *(v8bf*)(fxT + ((size_t)(b * HEADS + h) * DHEAD + d) * NTOK + n0 + 8 * half) = pk;
    }
  }
  __syncthreads();

  // ---- stage 2: slice logits = x_mid[h] @ W_slice, K = 64 per head ----
  for (int t = 0; t < 4; ++t) {
    const int tile = wave + 8 * t;           // 0..31
    const int h  = tile >> 2;
    const int gb = tile & 3;
    v8f acc = v8f_zero();
#pragma unroll
    for (int kk = 0; kk < DHEAD; kk += 32) {
      v16bf af = frag_a_rowmajor(&sm.a.xmid[0][h * DHEAD] + kk, INNER, lane);
      v16bf bf = frag_b_colmajor(wslT + (size_t)(gb * 16) * DHEAD + kk, DHEAD, lane);
      acc = wmma_bf16(af, bf, acc);
    }
    const int g = gb * 16 + ni;
    const float bias = bsl[g];
#pragma unroll
    for (int r = 0; r < 8; ++r)
      logit[r + 8 * half][h * GSLICE + g] = acc[r] + bias;
  }
  __syncthreads();   // logits ready; xa/xmid dead -> sstage may be written

  // ---- softmax over g per (row, head); temperature clipped to [0.1, 5] ----
  if (tid < 16 * HEADS) {
    const int r = tid & 15;
    const int h = tid >> 4;
    const float inv_t = 1.0f / fminf(fmaxf(temperature[h], 0.1f), 5.0f);
    float mx = -3.4e38f;
    for (int g = 0; g < GSLICE; ++g)
      mx = fmaxf(mx, logit[r][h * GSLICE + g] * inv_t);
    float sum = 0.f;
    for (int g = 0; g < GSLICE; ++g)
      sum += __expf(logit[r][h * GSLICE + g] * inv_t - mx);
    const float inv_s = 1.0f / sum;
    __bf16* dstN = swN + (((size_t)b * HEADS + h) * NTOK + n0 + r) * GSLICE;
    for (int g8 = 0; g8 < GSLICE; g8 += 8) {
      v8bf pk;
#pragma unroll
      for (int j = 0; j < 8; ++j) {
        const int g = g8 + j;
        const __bf16 w =
            (__bf16)(__expf(logit[r][h * GSLICE + g] * inv_t - mx) * inv_s);
        pk[j] = w;
        sm.sstage[h * GSLICE + g][r] = w;
      }
      *(v8bf*)(dstN + g8) = pk;   // 16B packed store
    }
  }
  __syncthreads();

  // cooperative swT store: each row is 16 consecutive n = two b128 stores
  for (int row = tid; row < HEADS * GSLICE; row += 256) {
    const int h = row >> 6, g = row & 63;
    const v8bf* src = (const v8bf*)&sm.sstage[row][0];
    v8bf* dst = (v8bf*)(swT + ((size_t)(b * HEADS + h) * GSLICE + g) * NTOK + n0);
    dst[0] = src[0];
    dst[1] = src[1];
  }
}

// ---------------------------------------------------------------------------
// K2: slice_tokens[g][d] += sum_n sw[n][g] * fx[n][d]   (+ norm[g])
//   LDS-free: A-frags stream from swT [G][N], B-frags from fxT [D][N].
//   grid = (B*H, POOL_CHUNKS). 8 waves; tiles (wave) and (wave+8) share the
//   same d-block, so one B-fragment feeds two WMMAs.
// ---------------------------------------------------------------------------
__global__ __launch_bounds__(256)
void slice_pool_kernel(const __bf16* __restrict__ fxT,   // [B*H][D][N]
                       const __bf16* __restrict__ swT,   // [B*H][G][N]
                       float* __restrict__ acc_tok,      // [B*H][G][D] f32
                       float* __restrict__ acc_norm)     // [B*H][G]    f32
{
  const int bh     = blockIdx.x;
  const int nstart = blockIdx.y * POOL_SPAN;
  const int tid  = threadIdx.x;
  const int wave = tid >> 5;
  const int lane = tid & 31;
  const int half = lane >> 4;
  const int ni   = lane & 15;

  const __bf16* swb = swT + (size_t)bh * GSLICE * NTOK;
  const __bf16* fxb = fxT + (size_t)bh * DHEAD * NTOK;

  const int db = wave & 3;            // shared d-block for both tiles
  const int gA = (wave >> 2) * 16;    // g-block 0 or 16
  const int gB = gA + 32;             // g-block 32 or 48
  v8f accA = v8f_zero();
  v8f accB = v8f_zero();

  const __bf16* aAp = swb + (size_t)gA * NTOK + nstart;
  const __bf16* aBp = swb + (size_t)gB * NTOK + nstart;
  const __bf16* bp  = fxb + (size_t)(db * 16) * NTOK + nstart;

  for (int kk = 0; kk < POOL_SPAN; kk += 32) {
    v16bf bf = frag_b_colmajor(bp + kk, NTOK, lane);
    v16bf a0 = frag_a_rowmajor(aAp + kk, NTOK, lane);
    v16bf a1 = frag_a_rowmajor(aBp + kk, NTOK, lane);
    accA = wmma_bf16(a0, bf, accA);
    accB = wmma_bf16(a1, bf, accB);
    // pull the next K-chunks toward the WGP (global_prefetch_b8)
    __builtin_prefetch(bp + kk + 128, 0, 1);
    __builtin_prefetch(aAp + kk + 128, 0, 1);
    __builtin_prefetch(aBp + kk + 128, 0, 1);
  }

#pragma unroll
  for (int r = 0; r < 8; ++r) {
    const int d = db * 16 + ni;
    atomicAdd(&acc_tok[((size_t)bh * GSLICE + gA + r + 8 * half) * DHEAD + d], accA[r]);
    atomicAdd(&acc_tok[((size_t)bh * GSLICE + gB + r + 8 * half) * DHEAD + d], accB[r]);
  }

  // norm partial: one thread per g sums its contiguous swT row segment
  if (tid < GSLICE) {
    const __bf16* row = swb + (size_t)tid * NTOK + nstart;
    float s = 0.f;
    for (int i = 0; i < POOL_SPAN; i += 8) {
      v8bf v = *(const v8bf*)(row + i);
#pragma unroll
      for (int j = 0; j < 8; ++j) s += (float)v[j];
    }
    atomicAdd(&acc_norm[(size_t)bh * GSLICE + tid], s);
  }
}

// ---------------------------------------------------------------------------
// K3: tiny per-(b,h) attention over 64 slice tokens (f32, negligible cost).
//   writes out_slice TRANSPOSED as bf16 [B*H][D][G] for the scatter WMMA.
// ---------------------------------------------------------------------------
__global__ __launch_bounds__(64)
void slice_attn_kernel(const float* __restrict__ acc_tok,
                       const float* __restrict__ acc_norm,
                       const float* __restrict__ Wq,
                       const float* __restrict__ Wk,
                       const float* __restrict__ Wv,
                       __bf16* __restrict__ osT)
{
  __shared__ float qs[GSLICE][DHEAD];
  __shared__ float ks_[GSLICE][DHEAD];
  __shared__ float vs[GSLICE][DHEAD];
  const int bh = blockIdx.x;
  const int g  = threadIdx.x;

  float srow[DHEAD];
  const float inv_norm = 1.0f / (acc_norm[(size_t)bh * GSLICE + g] + 1e-5f);
  for (int d = 0; d < DHEAD; ++d)
    srow[d] = acc_tok[((size_t)bh * GSLICE + g) * DHEAD + d] * inv_norm;

  for (int d = 0; d < DHEAD; ++d) {
    float q = 0.f, k = 0.f, v = 0.f;
    for (int c = 0; c < DHEAD; ++c) {
      const float s = srow[c];
      q += s * Wq[c * DHEAD + d];
      k += s * Wk[c * DHEAD + d];
      v += s * Wv[c * DHEAD + d];
    }
    qs[g][d] = q; ks_[g][d] = k; vs[g][d] = v;
  }
  __syncthreads();

  float sc[GSLICE];
  float mx = -3.4e38f;
  for (int j = 0; j < GSLICE; ++j) {
    float s = 0.f;
    for (int d = 0; d < DHEAD; ++d) s += qs[g][d] * ks_[j][d];
    sc[j] = s * ATTN_SCALE;
    mx = fmaxf(mx, sc[j]);
  }
  float sum = 0.f;
  for (int j = 0; j < GSLICE; ++j) { sc[j] = __expf(sc[j] - mx); sum += sc[j]; }
  const float inv_s = 1.0f / sum;
  for (int d = 0; d < DHEAD; ++d) {
    float o = 0.f;
    for (int j = 0; j < GSLICE; ++j) o += sc[j] * vs[j][d];
    osT[((size_t)bh * DHEAD + d) * GSLICE + g] = (__bf16)(o * inv_s);
  }
}

// ---------------------------------------------------------------------------
// K4: out = (sw @ out_slice per head, concat heads) @ W_out + b_out
//   one workgroup = 16 rows of n for one batch. 8 waves, 256 threads.
// ---------------------------------------------------------------------------
__global__ __launch_bounds__(256)
void scatter_out_kernel(const __bf16* __restrict__ swN,  // [B*H][N][G]
                        const __bf16* __restrict__ osT,  // [B*H][D][G]
                        const __bf16* __restrict__ woT,  // [CDIM][INNER]
                        const float* __restrict__ bout,
                        float* __restrict__ out)
{
  __shared__ __bf16 omid[16][INNER];  // 16 KB

  const int b    = blockIdx.y;
  const int n0   = blockIdx.x * 16;
  const int tid  = threadIdx.x;
  const int wave = tid >> 5;
  const int lane = tid & 31;
  const int half = lane >> 4;
  const int ni   = lane & 15;

  // stage 1: per-head scatter, K = G = 64
  for (int t = 0; t < 4; ++t) {
    const int tile = wave + 8 * t;     // 0..31
    const int h = tile >> 2, db = tile & 3;
    const __bf16* swrow = swN + (((size_t)b * HEADS + h) * NTOK + n0) * GSLICE;
    const __bf16* osblk = osT + ((size_t)(b * HEADS + h) * DHEAD + db * 16) * GSLICE;
    v8f acc = v8f_zero();
#pragma unroll
    for (int kk = 0; kk < GSLICE; kk += 32) {
      v16bf af = frag_a_rowmajor(swrow + kk, GSLICE, lane);
      v16bf bf = frag_b_colmajor(osblk + kk, GSLICE, lane);
      acc = wmma_bf16(af, bf, acc);
    }
    const int col = h * DHEAD + db * 16 + ni;
#pragma unroll
    for (int r = 0; r < 8; ++r)
      omid[r + 8 * half][col] = (__bf16)acc[r];
  }
  __syncthreads();

  // stage 2: output projection, K = INNER = 512.
  // Both output tiles of this wave consume the SAME A-fragment stream from
  // omid -> fuse them in one K-loop: 1 A-frag feeds 2 WMMAs.
  {
    const int colA = wave * 16;
    const int colB = (wave + 8) * 16;
    v8f acc0 = v8f_zero();
    v8f acc1 = v8f_zero();
#pragma unroll
    for (int kk = 0; kk < INNER; kk += 32) {
      v16bf af  = frag_a_rowmajor(&omid[0][0] + kk, INNER, lane);
      v16bf bf0 = frag_b_colmajor(woT + (size_t)colA * INNER + kk, INNER, lane);
      v16bf bf1 = frag_b_colmajor(woT + (size_t)colB * INNER + kk, INNER, lane);
      acc0 = wmma_bf16(af, bf0, acc0);
      acc1 = wmma_bf16(af, bf1, acc1);
    }
    const float bias0 = bout[colA + ni];
    const float bias1 = bout[colB + ni];
#pragma unroll
    for (int r = 0; r < 8; ++r) {
      const size_t rowoff = ((size_t)b * NTOK + n0 + r + 8 * half) * CDIM;
      out[rowoff + colA + ni] = acc0[r] + bias0;
      out[rowoff + colB + ni] = acc1[r] + bias1;
    }
  }
}

// ---------------------------------------------------------------------------
extern "C" void kernel_launch(void* const* d_in, const int* in_sizes, int n_in,
                              void* d_out, int out_size, void* d_ws, size_t ws_size,
                              hipStream_t stream) {
  (void)in_sizes; (void)n_in; (void)out_size; (void)ws_size;
  const float* x     = (const float*)d_in[0];
  const float* W_fx  = (const float*)d_in[1];
  const float* b_fx  = (const float*)d_in[2];
  const float* W_x   = (const float*)d_in[3];
  const float* b_x   = (const float*)d_in[4];
  const float* W_sl  = (const float*)d_in[5];
  const float* b_sl  = (const float*)d_in[6];
  const float* temp  = (const float*)d_in[7];
  const float* Wq    = (const float*)d_in[8];
  const float* Wk    = (const float*)d_in[9];
  const float* Wv    = (const float*)d_in[10];
  const float* W_out = (const float*)d_in[11];
  const float* b_out = (const float*)d_in[12];
  float* out = (float*)d_out;

  // ---- workspace layout (256B aligned) ----
  char* ws = (char*)d_ws;
  size_t off = 0;
  auto take = [&](size_t bytes) {
    void* p = ws + off;
    off = (off + bytes + 255) & ~(size_t)255;
    return p;
  };
  __bf16* wfxT = (__bf16*)take((size_t)INNER * CDIM * 2);
  __bf16* wxT  = (__bf16*)take((size_t)INNER * CDIM * 2);
  __bf16* wslT = (__bf16*)take((size_t)GSLICE * DHEAD * 2);
  __bf16* woT  = (__bf16*)take((size_t)CDIM * INNER * 2);
  __bf16* fxT  = (__bf16*)take((size_t)BATCH * HEADS * DHEAD * NTOK * 2);   // 128 MB
  __bf16* swN  = (__bf16*)take((size_t)BATCH * HEADS * NTOK * GSLICE * 2);  // 128 MB
  __bf16* swT  = (__bf16*)take((size_t)BATCH * HEADS * GSLICE * NTOK * 2);  // 128 MB
  float* acc_tok  = (float*)take((size_t)BATCH * HEADS * GSLICE * DHEAD * 4);
  float* acc_norm = (float*)take((size_t)BATCH * HEADS * GSLICE * 4);
  __bf16* osT = (__bf16*)take((size_t)BATCH * HEADS * DHEAD * GSLICE * 2);

  // ---- prep: transpose-convert weights to bf16, zero accumulators ----
  convT_kernel<<<(CDIM * INNER + 255) / 256, 256, 0, stream>>>(W_fx, wfxT, CDIM, INNER);
  convT_kernel<<<(CDIM * INNER + 255) / 256, 256, 0, stream>>>(W_x,  wxT,  CDIM, INNER);
  convT_kernel<<<(DHEAD * GSLICE + 255) / 256, 256, 0, stream>>>(W_sl, wslT, DHEAD, GSLICE);
  convT_kernel<<<(INNER * CDIM + 255) / 256, 256, 0, stream>>>(W_out, woT, INNER, CDIM);
  {
    const int c1 = BATCH * HEADS * GSLICE * DHEAD;
    zero_kernel<<<(c1 + 255) / 256, 256, 0, stream>>>(acc_tok, c1);
    const int c2 = BATCH * HEADS * GSLICE;
    zero_kernel<<<(c2 + 255) / 256, 256, 0, stream>>>(acc_norm, c2);
  }

  // ---- main pipeline ----
  proj_softmax_kernel<<<dim3(NTOK / 16, BATCH), 256, 0, stream>>>(
      x, wfxT, wxT, wslT, b_fx, b_x, b_sl, temp, fxT, swN, swT);

  slice_pool_kernel<<<dim3(BATCH * HEADS, POOL_CHUNKS), 256, 0, stream>>>(
      fxT, swT, acc_tok, acc_norm);

  slice_attn_kernel<<<BATCH * HEADS, 64, 0, stream>>>(
      acc_tok, acc_norm, Wq, Wk, Wv, osT);

  scatter_out_kernel<<<dim3(NTOK / 16, BATCH), 256, 0, stream>>>(
      swN, osT, woT, b_out, out);
}
